// SimpleAttention_7550552507065
// MI455X (gfx1250) — matrix operand order repaired and checked by reading, not compiled, not run on hardware
//
#include <hip/hip_runtime.h>
#include <hip/hip_bf16.h>

// ---------------------------------------------------------------------------
// MI455X (gfx1250) fused attention block:
//   Q/K/V projections + causal SDPA + output projection, all matmuls on
//   v_wmma_f32_16x16x32_bf16 (bf16 inputs, f32 accumulate).
//   B tiles staged to LDS via global_load_async_to_lds_b128 (ASYNCcnt),
//   explicitly double-buffered (manual 2x unroll, ping-pong registers).
// ---------------------------------------------------------------------------

typedef __bf16 bf16_t;
typedef __attribute__((ext_vector_type(16))) __bf16 v16bf;
typedef __attribute__((ext_vector_type(8)))  __bf16 v8bf;
typedef __attribute__((ext_vector_type(8)))  float  v8f;

#define SEQ   512
#define DIM   2048
#define HEADS 16
#define HD    128   // DIM / HEADS

static __device__ inline v8f wmma_bf16(v16bf a, v16bf b, v8f c) {
    return __builtin_amdgcn_wmma_f32_16x16x32_bf16(
        /*neg_a=*/false, a, /*neg_b=*/false, b,
        /*c_mod=*/(short)0, c, /*reuse_a=*/false, /*reuse_b=*/false);
}

// Build a 16-element bf16 fragment from two contiguous 8-element (16B) runs.
static __device__ inline v16bf make_frag(const bf16_t* p0, const bf16_t* p1) {
    v8bf a = *(const v8bf*)p0;
    v8bf b = *(const v8bf*)p1;
    v16bf r;
#pragma unroll
    for (int i = 0; i < 8; ++i) { r[i] = a[i]; r[i + 8] = b[i]; }
    return r;
}

static __device__ inline v8f zero8() {
    v8f z;
#pragma unroll
    for (int i = 0; i < 8; ++i) z[i] = 0.0f;
    return z;
}

// Generic LDS pointer -> wave-relative LDS byte offset (low 32 bits of the
// generic address per the LDS aperture rules).
static __device__ inline unsigned lds_off(const void* p) {
    return (unsigned)(size_t)p;
}

// 16-byte async copy global -> LDS (tracked by ASYNCcnt).
static __device__ inline void async_copy_b128(unsigned lds, const void* g) {
    asm volatile("global_load_async_to_lds_b128 %0, %1, off"
                 :: "v"(lds), "v"((unsigned long long)(size_t)g)
                 : "memory");
}

static __device__ inline void wait_async0() {
    asm volatile("s_wait_asynccnt 0x0" ::: "memory");
}

// ---------------------------------------------------------------------------
// f32 -> bf16 cast (grid-stride)
// ---------------------------------------------------------------------------
__global__ void cast_f32_to_bf16(const float* __restrict__ src,
                                 bf16_t* __restrict__ dst, int n) {
    int i = blockIdx.x * blockDim.x + threadIdx.x;
    int stride = gridDim.x * blockDim.x;
    for (; i < n; i += stride) dst[i] = (bf16_t)src[i];
}

// ---------------------------------------------------------------------------
// Transpose V per head: Vt[(h*HD+d)*SEQ + s] = Vb[s*DIM + h*HD + d]
// ---------------------------------------------------------------------------
__global__ void transpose_v(const bf16_t* __restrict__ Vb,
                            bf16_t* __restrict__ Vt) {
    int idx = blockIdx.x * blockDim.x + threadIdx.x;   // over SEQ*DIM
    if (idx >= SEQ * DIM) return;
    int s = idx >> 11;        // / DIM
    int c = idx & (DIM - 1);  // h*HD + d
    Vt[(size_t)c * SEQ + s] = Vb[idx];
}

// ---------------------------------------------------------------------------
// GEMM: C[M][N] = A[M][K] * W[N][K]^T   (i.e. y = x @ W^T, nn.Linear layout)
// A, W bf16; C f32 or bf16. Block tile 128(M) x 64(N), 256 threads = 8 waves.
// The 64x32 B tile is shared by all 8 waves -> async-staged to LDS.
// Manual 2x unroll: LDS buffers and A fragments ping-pong with no register
// copies; each step's A loads are issued one WMMA-quad ahead.
// K must be a multiple of 64 (here K = 2048, nsteps = 64).
// ---------------------------------------------------------------------------
__global__ __launch_bounds__(256)
void gemm_bf16_wmma(const bf16_t* __restrict__ A,
                    const bf16_t* __restrict__ W,
                    void* __restrict__ Cout,
                    int M, int N, int K, int outF32) {
    __shared__ bf16_t Bs[2][64 * 32];   // 2 x 4 KB, [n][k] row-major

    const int t    = threadIdx.x;
    const int wave = t >> 5;
    const int lane = t & 31;
    const int ml   = lane & 15;
    const int hi   = lane >> 4;

    const int m0 = blockIdx.y * 128 + wave * 16;
    const int n0 = blockIdx.x * 64;

    // Per-thread staging slice: 8 bf16 = 16 B.  n = t/4, k = (t%4)*8.
    const int sn = t >> 2;
    const int sk = (t & 3) * 8;
    const bf16_t* gsrc = W + (size_t)(n0 + sn) * K + sk;
    const unsigned lb0 = lds_off(&Bs[0][sn * 32 + sk]);
    const unsigned lb1 = lds_off(&Bs[1][sn * 32 + sk]);

    v8f acc[4];
#pragma unroll
    for (int j = 0; j < 4; ++j) acc[j] = zero8();

    const bf16_t* arow = A + (size_t)(m0 + ml) * K + hi * 8;
    const int nsteps = K / 32;          // even (K % 64 == 0)
    const int last   = nsteps - 1;

    async_copy_b128(lb0, gsrc);                 // stage 0 of B
    v16bf af0 = make_frag(arow, arow + 16);     // A fragment, step 0
    v16bf af1;

    for (int i = 0; i < nsteps; i += 2) {
        // ---- even step: consume Bs[0] + af0 ----
        wait_async0();
        __syncthreads();
        async_copy_b128(lb1, gsrc + (size_t)(i + 1) * 32);   // stage i+1
        {   // A fragment for step i+1 (latency hides behind the quad)
            const bf16_t* ap = arow + (size_t)(i + 1) * 32;
            af1 = make_frag(ap, ap + 16);
            __builtin_prefetch(ap + 256, 0, 0);
        }
#pragma unroll
        for (int j = 0; j < 4; ++j) {
            const bf16_t* bp = &Bs[0][(j * 16 + ml) * 32 + hi * 8];
            acc[j] = wmma_bf16(af0, make_frag(bp, bp + 16), acc[j]);
        }

        // ---- odd step: consume Bs[1] + af1 ----
        wait_async0();
        __syncthreads();
        if (i + 2 < nsteps)
            async_copy_b128(lb0, gsrc + (size_t)(i + 2) * 32);  // stage i+2
        {   // A fragment for step i+2 (pointer clamped; no select needed)
            int inext = (i + 2 < last) ? (i + 2) : last;
            const bf16_t* ap = arow + (size_t)inext * 32;
            af0 = make_frag(ap, ap + 16);
        }
#pragma unroll
        for (int j = 0; j < 4; ++j) {
            const bf16_t* bp = &Bs[1][(j * 16 + ml) * 32 + hi * 8];
            acc[j] = wmma_bf16(af1, make_frag(bp, bp + 16), acc[j]);
        }
    }

#pragma unroll
    for (int j = 0; j < 4; ++j) {
#pragma unroll
        for (int r = 0; r < 8; ++r) {
            int row = m0 + r + 8 * hi;
            int col = n0 + j * 16 + ml;
            if (outF32)
                ((float*)Cout)[(size_t)row * N + col] = acc[j][r];
            else
                ((bf16_t*)Cout)[(size_t)row * N + col] = (bf16_t)acc[j][r];
        }
    }
}

// ---------------------------------------------------------------------------
// Fused causal attention for one (head, 16-query tile).
// Block = 128 threads = 4 waves. Q tile async-staged in LDS; Q fragments
// hoisted out of the key loop; scores + softmax in LDS; both matmuls on
// WMMA; early exit past the causal frontier.
// ---------------------------------------------------------------------------
__global__ __launch_bounds__(128)
void attn_wmma(const bf16_t* __restrict__ Qb,
               const bf16_t* __restrict__ Kb,
               const bf16_t* __restrict__ Vt,
               bf16_t* __restrict__ Ob) {
    __shared__ bf16_t Qs[16 * HD];    // 4 KB
    __shared__ float  Sc[16 * SEQ];   // 32 KB
    __shared__ bf16_t Ps[16 * SEQ];   // 16 KB
    __shared__ float  red[16 * 8];
    __shared__ float  rowsum[16];
    __shared__ float  rowmax[16];

    const int h  = blockIdx.y;
    const int q0 = blockIdx.x * 16;
    const int t  = threadIdx.x;

    // ---- async-stage Q tile: 16 rows x 128 dims, 32 B per thread ----
    {
        int r  = t >> 3;
        int c0 = (t & 7) * 16;
        const bf16_t* src = Qb + (size_t)(q0 + r) * DIM + h * HD + c0;
        unsigned lq = lds_off(&Qs[r * HD + c0]);
        async_copy_b128(lq,      src);
        async_copy_b128(lq + 16, src + 8);
    }
    wait_async0();
    __syncthreads();

    const int wave = t >> 5;
    const int lane = t & 31;
    const int ml   = lane & 15;
    const int hi   = lane >> 4;
    const int qend = q0 + 15;
    const float scale = 0.08838834764831845f;   // 1/sqrt(128)

    // Q fragments are key-tile invariant: load once, reuse for all 32 tiles.
    v16bf qf[4];
#pragma unroll
    for (int dk = 0; dk < 4; ++dk) {
        const bf16_t* ap = &Qs[ml * HD + dk * 32 + hi * 8];
        qf[dk] = make_frag(ap, ap + 16);
    }

    // ---- scores = Q @ K^T * scale + causal mask ----
    for (int kt = wave; kt < SEQ / 16; kt += 4) {
        const int kbase = kt * 16;
        const int kcol  = kbase + ml;
        if (kbase > qend) {   // tile fully masked
#pragma unroll
            for (int r = 0; r < 8; ++r)
                Sc[(r + 8 * hi) * SEQ + kcol] = -INFINITY;
            continue;
        }
        // Preload the K-fragment quad, then run the WMMA quad.
        const bf16_t* krow = Kb + (size_t)kcol * DIM + h * HD + hi * 8;
        v16bf kf[4];
#pragma unroll
        for (int dk = 0; dk < 4; ++dk)
            kf[dk] = make_frag(krow + dk * 32, krow + dk * 32 + 16);
        v8f acc = zero8();
#pragma unroll
        for (int dk = 0; dk < 4; ++dk)
            acc = wmma_bf16(qf[dk], kf[dk], acc);
#pragma unroll
        for (int r = 0; r < 8; ++r) {
            int qrow = q0 + r + 8 * hi;
            float v = acc[r] * scale;
            if (kcol > qrow) v = -INFINITY;
            Sc[(r + 8 * hi) * SEQ + kcol] = v;
        }
    }
    __syncthreads();

    // ---- softmax over keys (8 threads per query row) ----
    {
        int r = t >> 3, sub = t & 7;
        float m = -INFINITY;
        for (int c = sub * 64; c < sub * 64 + 64; ++c)
            m = fmaxf(m, Sc[r * SEQ + c]);
        red[r * 8 + sub] = m;
    }
    __syncthreads();
    if (t < 16) {
        float m = red[t * 8];
#pragma unroll
        for (int i = 1; i < 8; ++i) m = fmaxf(m, red[t * 8 + i]);
        rowmax[t] = m;
    }
    __syncthreads();
    {
        int r = t >> 3, sub = t & 7;
        float mr = rowmax[r];
        float s = 0.0f;
        for (int c = sub * 64; c < sub * 64 + 64; ++c) {
            float e = __expf(Sc[r * SEQ + c] - mr);
            s += e;
            Ps[r * SEQ + c] = (bf16_t)e;   // unnormalized; divide at the end
        }
        red[r * 8 + sub] = s;
    }
    __syncthreads();
    if (t < 16) {
        float s = 0.0f;
#pragma unroll
        for (int i = 0; i < 8; ++i) s += red[t * 8 + i];
        rowsum[t] = s;
    }
    __syncthreads();

    // ---- out = P @ V  (skip key chunks past the causal frontier) ----
    const int nch = (qend + 32) >> 5;       // ceil((qend+1)/32)
    const int nchunks = nch < 16 ? nch : 16;
    for (int j = wave; j < 8; j += 4) {     // 8 column tiles of head_dim
        const bf16_t* prow = &Ps[ml * SEQ + hi * 8];
        const bf16_t* vrow =
            Vt + ((size_t)h * HD + j * 16 + ml) * SEQ + hi * 8;
        v8f acc = zero8();
        for (int ks = 0; ks < nchunks; ++ks) {
            const bf16_t* ap = prow + ks * 32;
            const bf16_t* bp = vrow + ks * 32;
            v16bf af = make_frag(ap, ap + 16);
            v16bf bfrag = make_frag(bp, bp + 16);
            acc = wmma_bf16(af, bfrag, acc);
        }
#pragma unroll
        for (int r = 0; r < 8; ++r) {
            int qrow = r + 8 * hi;
            float o = acc[r] / rowsum[qrow];
            Ob[(size_t)(q0 + qrow) * DIM + h * HD + j * 16 + ml] = (bf16_t)o;
        }
    }
}

// ---------------------------------------------------------------------------
// Host-side orchestration
// ---------------------------------------------------------------------------
extern "C" void kernel_launch(void* const* d_in, const int* in_sizes, int n_in,
                              void* d_out, int out_size, void* d_ws, size_t ws_size,
                              hipStream_t stream) {
    (void)in_sizes; (void)n_in; (void)out_size; (void)ws_size;

    const float* x  = (const float*)d_in[0];
    const float* wq = (const float*)d_in[3];
    const float* wk = (const float*)d_in[4];
    const float* wv = (const float*)d_in[5];
    const float* wo = (const float*)d_in[6];
    // d_in[1] (input_pos = arange), d_in[2] (causal mask), caches: semantics
    // are baked into the fused kernels.

    const int XE = SEQ * DIM;   // 1,048,576
    const int WE = DIM * DIM;   // 4,194,304

    bf16_t* base = (bf16_t*)d_ws;
    bf16_t* xb  = base;
    bf16_t* wqb = xb  + XE;
    bf16_t* wkb = wqb + WE;
    bf16_t* wvb = wkb + WE;
    bf16_t* wob = wvb + WE;
    bf16_t* Qb  = wob + WE;
    bf16_t* Kb  = Qb  + XE;
    bf16_t* Vb  = Kb  + XE;
    bf16_t* Vtb = Vb  + XE;
    bf16_t* Ob  = Vtb + XE;

    // 1) casts
    cast_f32_to_bf16<<<1024, 256, 0, stream>>>(x,  xb,  XE);
    cast_f32_to_bf16<<<4096, 256, 0, stream>>>(wq, wqb, WE);
    cast_f32_to_bf16<<<4096, 256, 0, stream>>>(wk, wkb, WE);
    cast_f32_to_bf16<<<4096, 256, 0, stream>>>(wv, wvb, WE);
    cast_f32_to_bf16<<<4096, 256, 0, stream>>>(wo, wob, WE);

    // 2) Q/K/V projections (bf16 out)
    dim3 gg(DIM / 64, SEQ / 128);   // (32, 4)
    gemm_bf16_wmma<<<gg, 256, 0, stream>>>(xb, wqb, Qb, SEQ, DIM, DIM, 0);
    gemm_bf16_wmma<<<gg, 256, 0, stream>>>(xb, wkb, Kb, SEQ, DIM, DIM, 0);
    gemm_bf16_wmma<<<gg, 256, 0, stream>>>(xb, wvb, Vb, SEQ, DIM, DIM, 0);

    // 3) V transpose for P@V fragment layout
    transpose_v<<<(SEQ * DIM + 255) / 256, 256, 0, stream>>>(Vb, Vtb);

    // 4) fused causal attention
    dim3 ga(SEQ / 16, HEADS);       // (32, 16)
    attn_wmma<<<ga, 128, 0, stream>>>(Qb, Kb, Vtb, Ob);

    // 5) output projection (f32 out -> d_out)
    gemm_bf16_wmma<<<gg, 256, 0, stream>>>(Ob, wob, d_out, SEQ, DIM, DIM, 1);
}
